// MixtralMoE_34668976013446
// MI455X (gfx1250) — matrix operand order repaired and checked by reading, not compiled, not run on hardware
//
#include <hip/hip_runtime.h>
#include <hip/hip_bf16.h>
#include <math.h>

// ---------------- problem constants ----------------
#define T_TOK   2048      // 2*1024 tokens
#define DHID    2048      // HIDDEN
#define DFFN    7168      // FFN
#define NEXP    8
#define TOPK    2

// WMMA vector types (CDNA5, wave32)
typedef __bf16 bf16_t;
typedef bf16_t v16bf __attribute__((ext_vector_type(16)));
typedef float  v8f   __attribute__((ext_vector_type(8)));

struct B16 { unsigned int w0, w1, w2, w3; };          // trivially-copyable 16B chunk
union  Frag { bf16_t e[16]; B16 b[2]; v16bf v; };

// native fp32 -> bf16 (clang lowers fptrunc-to-bfloat to gfx1250 cvt ops)
static __device__ __forceinline__ bf16_t f2bf(float f) { return (bf16_t)f; }

static __device__ __forceinline__ void atomic_fadd(float* p, float v) {
    __hip_atomic_fetch_add(p, v, __ATOMIC_RELAXED, __HIP_MEMORY_SCOPE_AGENT);
}

// fast sigmoid via hardware v_rcp_f32 (activation precision is fine in bf16 land)
static __device__ __forceinline__ float fast_sigmoid(float x) {
    return __builtin_amdgcn_rcpf(1.0f + __expf(-x));
}

// ---------------- kernel 0: zero y + counters ----------------
__global__ __launch_bounds__(256) void moe_zero_kernel(float* __restrict__ y, int n,
                                                       int* __restrict__ cnt) {
    for (int i = blockIdx.x * blockDim.x + threadIdx.x; i < n; i += gridDim.x * blockDim.x)
        y[i] = 0.0f;
    if (blockIdx.x == 0 && threadIdx.x < NEXP) cnt[threadIdx.x] = 0;
}

// ---------------- kernel 1: router (wave per token) ----------------
__global__ __launch_bounds__(128) void moe_router_kernel(
        const float* __restrict__ h, const float* __restrict__ wg,
        float* __restrict__ logits_out,
        int* __restrict__ cnt, int* __restrict__ list, float* __restrict__ wts) {
    const int lane = threadIdx.x & 31;
    const int t    = blockIdx.x * 4 + (threadIdx.x >> 5);
    if (t >= T_TOK) return;

    float logit[NEXP];
    const float* hrow = h + (size_t)t * DHID;
#pragma unroll
    for (int e = 0; e < NEXP; ++e) {
        const float* grow = wg + (size_t)e * DHID;
        float acc = 0.0f;
        for (int j = lane; j < DHID; j += 32)
            acc = fmaf(hrow[j], grow[j], acc);
        // wave32 butterfly reduction -> all lanes hold the sum
#pragma unroll
        for (int off = 16; off > 0; off >>= 1)
            acc += __shfl_xor(acc, off, 32);
        logit[e] = acc;
    }

    if (lane == 0) {
        float* lo = logits_out + (size_t)t * NEXP;
#pragma unroll
        for (int e = 0; e < NEXP; ++e) lo[e] = logit[e];
        // top-2 (descending, lower index wins ties — matches lax.top_k)
        int i1 = 0;
#pragma unroll
        for (int e = 1; e < NEXP; ++e) if (logit[e] > logit[i1]) i1 = e;
        int i2 = (i1 == 0) ? 1 : 0;
#pragma unroll
        for (int e = 0; e < NEXP; ++e)
            if (e != i1 && logit[e] > logit[i2]) i2 = e;
        // softmax over the two selected logits
        float m  = logit[i1];
        float e2 = __expf(logit[i2] - m);
        float inv = 1.0f / (1.0f + e2);
        float w1v = inv, w2v = e2 * inv;

        int s1 = __hip_atomic_fetch_add(&cnt[i1], 1, __ATOMIC_RELAXED, __HIP_MEMORY_SCOPE_AGENT);
        list[i1 * T_TOK + s1] = t;  wts[i1 * T_TOK + s1] = w1v;
        int s2 = __hip_atomic_fetch_add(&cnt[i2], 1, __ATOMIC_RELAXED, __HIP_MEMORY_SCOPE_AGENT);
        list[i2 * T_TOK + s2] = t;  wts[i2 * T_TOK + s2] = w2v;
    }
}

// ---------------- kernel 2: expert SwiGLU + down-proj (WMMA bf16) ----------------
// grid = (128 tile-slots, 8 experts); block = 256 (8 waves). One workgroup handles
// one expert x 16-token tile. Full 16x7168 bf16 activation lives in LDS (229 KB —
// only possible with CDNA5's 320 KB WGP LDS).
__global__ __launch_bounds__(256) void moe_expert_kernel(
        const float* __restrict__ h,
        const float* __restrict__ w1, const float* __restrict__ w3,
        const float* __restrict__ w2,
        const int* __restrict__ cnt, const int* __restrict__ list,
        const float* __restrict__ wts,
        float* __restrict__ y) {
    extern __shared__ char smem[];
    bf16_t* hbuf = (bf16_t*)smem;                 // [16][DHID] bf16  (64 KB)
    bf16_t* abuf = hbuf + 16 * DHID;              // [16][DFFN] bf16  (224 KB)
    int*   trow = (int*)(abuf + 16 * DFFN);       // [16] token rows
    float* twt  = (float*)(trow + 16);            // [16] routing weights

    const int e   = blockIdx.y;
    const int tb  = blockIdx.x;
    const int n_e = cnt[e];
    if (tb * 16 >= n_e) return;                   // uniform early exit

    const int tid  = threadIdx.x;
    const int lane = tid & 31;
    const int wave = tid >> 5;

    // stage token ids / weights for this 16-token tile (pad -> weight 0)
    if (tid < 16) {
        int slot = tb * 16 + tid;
        if (slot < n_e) { trow[tid] = list[e * T_TOK + slot]; twt[tid] = wts[e * T_TOK + slot]; }
        else            { trow[tid] = list[e * T_TOK + tb * 16]; twt[tid] = 0.0f; }
    }
    __syncthreads();

    // cooperative gather of 16 token rows -> LDS as bf16 (pairs for packed cvt)
    for (int i = tid * 2; i < 16 * DHID; i += 512) {
        int r = i >> 11, c = i & (DHID - 1);
        const float* src = h + (size_t)trow[r] * DHID + c;
        hbuf[i]     = f2bf(src[0]);
        hbuf[i + 1] = f2bf(src[1]);
    }
    __syncthreads();

    const float* w1e = w1 + (size_t)e * DHID * DFFN;
    const float* w3e = w3 + (size_t)e * DHID * DFFN;
    const float* w2e = w2 + (size_t)e * DFFN * DHID;

    const int m     = lane & 15;
    const int koffA = (lane < 16) ? 0 : 16;   // A-frag K half (contiguous, matches B)
    const int koffB = (lane < 16) ? 0 : 16;   // B-frag K half
    const int rbase = (lane < 16) ? 0 : 8;    // C/D row base per half-wave

    // -------- phase 1: g = h@w1, u = h@w3, a = silu(g)*u --------
    for (int nt = wave; nt < DFFN / 16; nt += 8) {
        const int col = nt * 16 + m;
        v8f accg = {}, accu = {};
#pragma unroll 2
        for (int kb = 0; kb < DHID; kb += 32) {
            Frag af;
            const bf16_t* ap = hbuf + m * DHID + kb + koffA;
            af.b[0] = *(const B16*)(ap);
            af.b[1] = *(const B16*)(ap + 8);

            const float* p1 = w1e + (size_t)(kb + koffB) * DFFN + col;
            const float* p3 = w3e + (size_t)(kb + koffB) * DFFN + col;
            __builtin_prefetch(p1 + (size_t)64 * DFFN, 0, 0);   // global_prefetch_b8
            __builtin_prefetch(p3 + (size_t)64 * DFFN, 0, 0);
            Frag bg, bu;
#pragma unroll
            for (int i = 0; i < 16; ++i) {
                bg.e[i] = f2bf(p1[(size_t)i * DFFN]);
                bu.e[i] = f2bf(p3[(size_t)i * DFFN]);
            }
            accg = __builtin_amdgcn_wmma_f32_16x16x32_bf16(false, af.v, false, bg.v,
                                                           (short)0, accg, false, false);
            accu = __builtin_amdgcn_wmma_f32_16x16x32_bf16(false, af.v, false, bu.v,
                                                           (short)0, accu, false, false);
        }
#pragma unroll
        for (int r = 0; r < 8; ++r) {
            float g = accg[r], u = accu[r];
            float a = g * fast_sigmoid(g) * u;                  // silu(g)*u via v_rcp_f32
            abuf[(size_t)(r + rbase) * DFFN + col] = f2bf(a);
        }
    }
    __syncthreads();

    // -------- phase 2: o = a @ w2, scaled combine into y --------
    float wrow[8]; int trl[8];
#pragma unroll
    for (int r = 0; r < 8; ++r) { wrow[r] = twt[r + rbase]; trl[r] = trow[r + rbase]; }

    for (int ot = wave; ot < DHID / 16; ot += 8) {
        const int col = ot * 16 + m;
        v8f acc = {};
#pragma unroll 2
        for (int kb = 0; kb < DFFN; kb += 32) {
            Frag af;
            const bf16_t* ap = abuf + (size_t)m * DFFN + kb + koffA;
            af.b[0] = *(const B16*)(ap);
            af.b[1] = *(const B16*)(ap + 8);

            const float* p2 = w2e + (size_t)(kb + koffB) * DHID + col;
            __builtin_prefetch(p2 + (size_t)64 * DHID, 0, 0);
            Frag bw;
#pragma unroll
            for (int i = 0; i < 16; ++i)
                bw.e[i] = f2bf(p2[(size_t)i * DHID]);
            acc = __builtin_amdgcn_wmma_f32_16x16x32_bf16(false, af.v, false, bw.v,
                                                          (short)0, acc, false, false);
        }
#pragma unroll
        for (int r = 0; r < 8; ++r) {
            float v = acc[r] * wrow[r];
            atomic_fadd(&y[(size_t)trl[r] * DHID + col], v);
        }
    }
}

// ---------------- launch ----------------
extern "C" void kernel_launch(void* const* d_in, const int* in_sizes, int n_in,
                              void* d_out, int out_size, void* d_ws, size_t ws_size,
                              hipStream_t stream) {
    const float* h  = (const float*)d_in[0];
    const float* wg = (const float*)d_in[1];
    const float* w1 = (const float*)d_in[2];
    const float* w3 = (const float*)d_in[3];
    const float* w2 = (const float*)d_in[4];

    float* y      = (float*)d_out;                       // [T, DHID]
    float* logits = y + (size_t)T_TOK * DHID;            // [T, NEXP]

    int*   cnt  = (int*)d_ws;                            // [8]
    int*   list = cnt + 8;                               // [8][T]
    float* wts  = (float*)(list + NEXP * T_TOK);         // [8][T]

    // 0) zero y + expert counters
    moe_zero_kernel<<<2048, 256, 0, stream>>>(y, T_TOK * DHID, cnt);

    // 1) router: logits, top-2 softmax, per-expert token lists
    moe_router_kernel<<<T_TOK / 4, 128, 0, stream>>>(h, wg, logits, cnt, list, wts);

    // 2) expert FFN: fixed grid (graph-capture safe), early-exit on empty tiles
    const size_t smem = (size_t)16 * DHID * 2 + (size_t)16 * DFFN * 2 + 16 * 4 + 16 * 4;
    dim3 grid(128, NEXP, 1);
    moe_expert_kernel<<<grid, 256, smem, stream>>>(h, w1, w3, w2, cnt, list, wts, y);
}
